// SelfAttention_9491877724198
// MI455X (gfx1250) — matrix operand order repaired and checked by reading, not compiled, not run on hardware
//
#include <hip/hip_runtime.h>
#include <hip/hip_bf16.h>
#include <math.h>

#define D_MODEL 2048
#define N_HEAD  16
#define HEAD_DIM 128
#define BATCH   2
#define SEQ     2048
#define M_TOT   (BATCH*SEQ)   // 4096

typedef __attribute__((ext_vector_type(16))) __bf16 v16bf;
typedef __attribute__((ext_vector_type(8)))  __bf16 v8bf;
typedef __attribute__((ext_vector_type(8)))  float  v8f;

static __device__ __forceinline__ v16bf mk_v16(v8bf lo, v8bf hi) {
  v16bf r;
#pragma unroll
  for (int i = 0; i < 8; ++i) { r[i] = lo[i]; r[8 + i] = hi[i]; }
  return r;
}
static __device__ __forceinline__ v8bf ld8(const __bf16* p) {
  return *(const v8bf*)p;
}

// ---------------------------------------------------------------------------
// CDNA5 async global->LDS staging (ASYNCcnt path) via inline asm:
//   global_load_async_to_lds_b128 vdst(LDS byte addr), vaddr(64b global), off
// 16 bytes per lane, tracked by ASYNCcnt; ordered with s_wait_asynccnt.
// ---------------------------------------------------------------------------
static __device__ __forceinline__ void async_copy16(__bf16* l, const __bf16* g) {
  unsigned lds = (unsigned)(size_t)(__attribute__((address_space(3))) __bf16*)l;
  unsigned long long ga = (unsigned long long)(size_t)g;
  asm volatile("global_load_async_to_lds_b128 %0, %1, off"
               :: "v"(lds), "v"(ga) : "memory");
}

static __device__ __forceinline__ void wait_async0() {
#if defined(__AMDGCN__) && __has_builtin(__builtin_amdgcn_s_wait_asynccnt)
  __builtin_amdgcn_s_wait_asynccnt(0);
#else
  asm volatile("s_wait_asynccnt 0x0" ::: "memory");
#endif
}

// ---------------------------------------------------------------------------
// f32 -> bf16 conversion (grid-stride)
// ---------------------------------------------------------------------------
__global__ __launch_bounds__(256) void cvt_f32_bf16(const float* __restrict__ in,
                                                    __bf16* __restrict__ out, int n) {
  int i = blockIdx.x * 256 + threadIdx.x;
  int stride = gridDim.x * 256;
  for (; i < n; i += stride) out[i] = (__bf16)in[i];
}

// f32 [K][N] -> bf16 transposed [N][K]  (one-time weight pass)
__global__ __launch_bounds__(256) void cvt_transpose_bf16(
    const float* __restrict__ in, __bf16* __restrict__ out, int K, int N) {
  int tx = threadIdx.x & 15, ty = threadIdx.x >> 4;
  int n = blockIdx.x * 16 + tx;
  int k = blockIdx.y * 16 + ty;
  out[(size_t)n * K + k] = (__bf16)in[(size_t)k * N + n];
}

// ---------------------------------------------------------------------------
// bf16 GEMM, f32 accumulate + bias:  C[M,N] = A[M,K] * B[K,N] + bias
// B given pre-transposed BT[N][K].  Block tile 128x128, 8 waves -> 32x64 each,
// K-step 32, double-buffered async global->LDS staging.
// ---------------------------------------------------------------------------
template <bool OUT_BF16>
__global__ __launch_bounds__(256) void hgemm_bias(
    const __bf16* __restrict__ A, const __bf16* __restrict__ BT,
    const float* __restrict__ bias,
    __bf16* __restrict__ outB, float* __restrict__ outF,
    int M, int N, int K) {
  __shared__ __align__(16) __bf16 As[2][128 * 32];  // [row][k]
  __shared__ __align__(16) __bf16 Bt[2][128 * 32];  // [n][k]

  const int tid = threadIdx.x;
  const int wid = tid >> 5, ln = tid & 31;
  const int lc = ln & 15;
  const int hi8  = (ln < 16) ? 0 : 8;
  const int hi16 = (ln < 16) ? 0 : 16;
  const int bm0 = blockIdx.y * 128;
  const int bn0 = blockIdx.x * 128;
  const int wm = (wid & 3) * 32;
  const int wn = (wid >> 2) * 64;

  const int chR = tid >> 2;            // 0..63 -> two rows per thread pass
  const int chK = (tid & 3) * 8;       // 16B chunk within 32-wide k-step

  v8f acc[2][4];
#pragma unroll
  for (int mi = 0; mi < 2; ++mi)
#pragma unroll
    for (int ni = 0; ni < 4; ++ni) acc[mi][ni] = {};

  // stage one 128x32 A tile + 128x32 B^T tile (contiguous 16B chunks)
  auto stage = [&](int buf, int k0) {
#pragma unroll
    for (int it = 0; it < 2; ++it) {
      int r = chR + it * 64;
      async_copy16(&As[buf][r * 32 + chK],
                   &A[(size_t)(bm0 + r) * K + k0 + chK]);
      async_copy16(&Bt[buf][r * 32 + chK],
                   &BT[(size_t)(bn0 + r) * K + k0 + chK]);
    }
  };

  const int nk = K >> 5;
  stage(0, 0);
  for (int i = 0; i < nk; ++i) {
    wait_async0();
    __syncthreads();
    if (i + 1 < nk) {
      stage((i + 1) & 1, (i + 1) << 5);
      if (i + 2 < nk) {   // warm L2 for the tile after next
        __builtin_prefetch(&A[(size_t)(bm0 + chR) * K + ((i + 2) << 5) + chK], 0, 1);
        __builtin_prefetch(&BT[(size_t)(bn0 + chR) * K + ((i + 2) << 5) + chK], 0, 1);
      }
    }
    const int buf = i & 1;

    v16bf af[2], bfm[4];
#pragma unroll
    for (int mi = 0; mi < 2; ++mi) {
      const __bf16* pr = &As[buf][(wm + mi * 16 + lc) * 32];
      af[mi] = mk_v16(ld8(pr + hi8), ld8(pr + 16 + hi8));
    }
#pragma unroll
    for (int ni = 0; ni < 4; ++ni) {
      const __bf16* pr = &Bt[buf][(wn + ni * 16 + lc) * 32];
      bfm[ni] = mk_v16(ld8(pr + hi8), ld8(pr + 16 + hi8));
    }
#pragma unroll
    for (int mi = 0; mi < 2; ++mi)
#pragma unroll
      for (int ni = 0; ni < 4; ++ni)
        acc[mi][ni] = __builtin_amdgcn_wmma_f32_16x16x32_bf16(
            false, af[mi], false, bfm[ni], (short)0, acc[mi][ni], false, false);
  }

#pragma unroll
  for (int mi = 0; mi < 2; ++mi)
#pragma unroll
    for (int ni = 0; ni < 4; ++ni) {
      int col = bn0 + wn + ni * 16 + lc;
      float bv = bias ? bias[col] : 0.f;
#pragma unroll
      for (int r = 0; r < 8; ++r) {
        int row = bm0 + wm + mi * 16 + hi8 + r;
        float v = acc[mi][ni][r] + bv;
        if (OUT_BF16) outB[(size_t)row * N + col] = (__bf16)v;
        else          outF[(size_t)row * N + col] = v;
      }
    }
}

// ---------------------------------------------------------------------------
// RoPE + split qkv -> Q/K per-head [B,H,T,HD] bf16 (1/sqrt(hd) folded into Q)
// and V per-head TRANSPOSED [B,H,HD,T] bf16 so attention can async-stage it.
// one thread per (b,h,t,i), i in [0,64)
// ---------------------------------------------------------------------------
__global__ __launch_bounds__(256) void rope_split(
    const __bf16* __restrict__ qkv, __bf16* __restrict__ qd,
    __bf16* __restrict__ kd, __bf16* __restrict__ vT) {
  int idx = blockIdx.x * 256 + threadIdx.x;   // B*H*T*64 = 2^22 threads
  int i = idx & 63;
  int t = (idx >> 6) & (SEQ - 1);
  int h = (idx >> 17) & (N_HEAD - 1);
  int b = idx >> 21;
  int bh = b * N_HEAD + h;
  size_t src = (size_t)(b * SEQ + t) * (3 * D_MODEL) + h * HEAD_DIM;
  size_t dst = ((size_t)bh * SEQ + t) * HEAD_DIM;

  float rate = __expf(-0.14391156831212787f * (float)i);  // 10000^(-2i/128)
  float th = (float)t * rate;
  float c = __cosf(th), s = __sinf(th);
  const float qs = 0.08838834764831845f;                  // 1/sqrt(128)

  float q1 = (float)qkv[src + i],           q2 = (float)qkv[src + 64 + i];
  float k1 = (float)qkv[src + D_MODEL + i], k2 = (float)qkv[src + D_MODEL + 64 + i];
  qd[dst + i]      = (__bf16)((q1 * c - q2 * s) * qs);
  qd[dst + 64 + i] = (__bf16)((q1 * s + q2 * c) * qs);
  kd[dst + i]      = (__bf16)(k1 * c - k2 * s);
  kd[dst + 64 + i] = (__bf16)(k1 * s + k2 * c);
  // V transposed: vT[bh][d][t]
  vT[((size_t)bh * HEAD_DIM + i)      * SEQ + t] = qkv[src + 2 * D_MODEL + i];
  vT[((size_t)bh * HEAD_DIM + i + 64) * SEQ + t] = qkv[src + 2 * D_MODEL + 64 + i];
}

// ---------------------------------------------------------------------------
// Flash attention: block = (head, 128 query rows), 8 waves x 16-row strips.
// 64-key blocks, online softmax, double-buffered async K/V staging.
// ---------------------------------------------------------------------------
__global__ __launch_bounds__(256) void flash_attn(
    const __bf16* __restrict__ Q, const __bf16* __restrict__ Kh,
    const __bf16* __restrict__ Vt_g, __bf16* __restrict__ Y) {
  __shared__ __align__(16) __bf16 Qs[128 * 128];     // [qrow][d]
  __shared__ __align__(16) __bf16 Ks[2][64 * 128];   // [key][d]
  __shared__ __align__(16) __bf16 Vt[2][128 * 64];   // [d][key]
  __shared__ __align__(16) __bf16 Ps[128 * 64];      // [qrow][key]

  const int tid = threadIdx.x;
  const int wid = tid >> 5, ln = tid & 31;
  const int lc = ln & 15;
  const int hi8  = (ln < 16) ? 0 : 8;
  const int hi16 = (ln < 16) ? 0 : 16;
  const int bh = blockIdx.x;           // b*H + h
  const int qb = blockIdx.y;
  const int q0 = qb * 128;
  const size_t headBase = (size_t)bh * SEQ * HEAD_DIM;   // Q/K base
  const size_t vBase    = (size_t)bh * HEAD_DIM * SEQ;   // V^T base

  // stage K (row-major) and V^T (d-major) tiles for key block j
  auto stage_kv = [&](int buf, int j) {
#pragma unroll
    for (int it = 0; it < 4; ++it) {
      int ch = tid + it * 256;
      {  // K: 64 rows x 128 d  -> 1024 chunks
        int r = ch >> 4, d8 = (ch & 15) * 8;
        async_copy16(&Ks[buf][r * 128 + d8],
                     &Kh[headBase + (size_t)(j * 64 + r) * HEAD_DIM + d8]);
      }
      {  // V^T: 128 d x 64 keys -> 1024 chunks
        int d = ch >> 3, t8 = (ch & 7) * 8;
        async_copy16(&Vt[buf][d * 64 + t8],
                     &Vt_g[vBase + (size_t)d * SEQ + j * 64 + t8]);
      }
    }
  };

  // stage Q tile 128x128 + first K/V block
#pragma unroll
  for (int it = 0; it < 8; ++it) {
    int ch = tid + it * 256;
    int r = ch >> 4, d8 = (ch & 15) * 8;
    async_copy16(&Qs[r * 128 + d8],
                 &Q[headBase + (size_t)(q0 + r) * HEAD_DIM + d8]);
  }
  stage_kv(0, 0);
  wait_async0();
  __syncthreads();

  v16bf aQ[4];                          // wave's 16 query rows, K=128
#pragma unroll
  for (int kk = 0; kk < 4; ++kk) {
    const __bf16* pr = &Qs[(wid * 16 + lc) * 128 + kk * 32];
    aQ[kk] = mk_v16(ld8(pr + hi8), ld8(pr + 16 + hi8));
  }

  v8f o[8];
#pragma unroll
  for (int nt = 0; nt < 8; ++nt) o[nt] = {};
  float mrun[8], lrun[8];
#pragma unroll
  for (int r = 0; r < 8; ++r) { mrun[r] = -1e30f; lrun[r] = 0.f; }

  const int jmax = 2 * qb + 1;
  for (int j = 0; j <= jmax; ++j) {
    if (j) {                            // tile j was staged last iteration
      wait_async0();
      __syncthreads();
    }
    if (j < jmax) stage_kv((j + 1) & 1, j + 1);   // overlap with compute
    const int buf = j & 1;

    v8f s[4];                           // S strip 16x64
#pragma unroll
    for (int nt = 0; nt < 4; ++nt) {
      s[nt] = {};
#pragma unroll
      for (int kk = 0; kk < 4; ++kk) {
        const __bf16* pr = &Ks[buf][(nt * 16 + lc) * 128 + kk * 32];
        v16bf bK = mk_v16(ld8(pr + hi8), ld8(pr + 16 + hi8));
        s[nt] = __builtin_amdgcn_wmma_f32_16x16x32_bf16(
            false, aQ[kk], false, bK, (short)0, s[nt], false, false);
      }
    }

    const int rowg0 = q0 + wid * 16 + hi8;   // causal mask
#pragma unroll
    for (int nt = 0; nt < 4; ++nt) {
      int colg = j * 64 + nt * 16 + lc;
#pragma unroll
      for (int r = 0; r < 8; ++r)
        if (colg > rowg0 + r) s[nt][r] = -1e30f;
    }

    float alpha[8];                     // online softmax per row slot
#pragma unroll
    for (int r = 0; r < 8; ++r) {
      float mx = s[0][r];
#pragma unroll
      for (int nt = 1; nt < 4; ++nt) mx = fmaxf(mx, s[nt][r]);
#pragma unroll
      for (int off = 8; off; off >>= 1) mx = fmaxf(mx, __shfl_xor(mx, off, 32));
      float mnew = fmaxf(mrun[r], mx);
      alpha[r] = __expf(mrun[r] - mnew);
      mrun[r] = mnew;
      float rs = 0.f;
#pragma unroll
      for (int nt = 0; nt < 4; ++nt) {
        float p = __expf(s[nt][r] - mnew);
        s[nt][r] = p;
        rs += p;
      }
#pragma unroll
      for (int off = 8; off; off >>= 1) rs += __shfl_xor(rs, off, 32);
      lrun[r] = lrun[r] * alpha[r] + rs;
    }

#pragma unroll
    for (int nt = 0; nt < 8; ++nt)      // rescale O
#pragma unroll
      for (int r = 0; r < 8; ++r) o[nt][r] *= alpha[r];

#pragma unroll
    for (int nt = 0; nt < 4; ++nt)      // P -> LDS (C-layout -> A-layout)
#pragma unroll
      for (int r = 0; r < 8; ++r)
        Ps[(wid * 16 + hi8 + r) * 64 + nt * 16 + lc] = (__bf16)s[nt][r];

    asm volatile("s_wait_dscnt 0x0" ::: "memory");

#pragma unroll
    for (int kk = 0; kk < 2; ++kk) {    // O += P * V
      const __bf16* pp = &Ps[(wid * 16 + lc) * 64 + kk * 32];
      v16bf aP = mk_v16(ld8(pp + hi8), ld8(pp + 16 + hi8));
#pragma unroll
      for (int nt = 0; nt < 8; ++nt) {
        const __bf16* pv = &Vt[buf][(nt * 16 + lc) * 64 + kk * 32];
        v16bf bV = mk_v16(ld8(pv + hi8), ld8(pv + 16 + hi8));
        o[nt] = __builtin_amdgcn_wmma_f32_16x16x32_bf16(
            false, aP, false, bV, (short)0, o[nt], false, false);
      }
    }
  }

  const int b = bh / N_HEAD, h = bh % N_HEAD;   // write y [B*T, D] bf16
#pragma unroll
  for (int r = 0; r < 8; ++r) {
    float inv = 1.0f / lrun[r];
    size_t row = (size_t)(b * SEQ + q0 + wid * 16 + hi8 + r);
#pragma unroll
    for (int nt = 0; nt < 8; ++nt)
      Y[row * D_MODEL + h * HEAD_DIM + nt * 16 + lc] =
          (__bf16)(o[nt][r] * inv);
  }
}

// ---------------------------------------------------------------------------
// launch
// ---------------------------------------------------------------------------
extern "C" void kernel_launch(void* const* d_in, const int* in_sizes, int n_in,
                              void* d_out, int out_size, void* d_ws, size_t ws_size,
                              hipStream_t stream) {
  const float* x     = (const float*)d_in[0];
  // d_in[1] = mask (causal, handled analytically)
  const float* Wqkv  = (const float*)d_in[2];
  const float* bqkv  = (const float*)d_in[3];
  const float* Wproj = (const float*)d_in[4];
  const float* bproj = (const float*)d_in[5];
  float* out = (float*)d_out;

  char* p = (char*)d_ws;
  __bf16* x_bf      = (__bf16*)p; p += (size_t)M_TOT * D_MODEL * 2;
  __bf16* wqkv_t    = (__bf16*)p; p += (size_t)D_MODEL * 3 * D_MODEL * 2;   // [3D][D]
  __bf16* wproj_t   = (__bf16*)p; p += (size_t)D_MODEL * D_MODEL * 2;       // [D][D]
  __bf16* qkv_bf    = (__bf16*)p; p += (size_t)M_TOT * 3 * D_MODEL * 2;
  __bf16* q_r       = (__bf16*)p; p += (size_t)M_TOT * D_MODEL * 2;
  __bf16* k_r       = (__bf16*)p; p += (size_t)M_TOT * D_MODEL * 2;
  __bf16* v_t       = (__bf16*)p; p += (size_t)M_TOT * D_MODEL * 2;         // [B,H,HD,T]
  __bf16* y_bf      = (__bf16*)p; p += (size_t)M_TOT * D_MODEL * 2;

  cvt_f32_bf16<<<2048, 256, 0, stream>>>(x, x_bf, M_TOT * D_MODEL);
  cvt_transpose_bf16<<<dim3(3 * D_MODEL / 16, D_MODEL / 16), 256, 0, stream>>>(
      Wqkv, wqkv_t, D_MODEL, 3 * D_MODEL);
  cvt_transpose_bf16<<<dim3(D_MODEL / 16, D_MODEL / 16), 256, 0, stream>>>(
      Wproj, wproj_t, D_MODEL, D_MODEL);

  // qkv = x @ Wqkv + bqkv   (4096 x 6144 x 2048)
  hgemm_bias<true><<<dim3(3 * D_MODEL / 128, M_TOT / 128), 256, 0, stream>>>(
      x_bf, wqkv_t, bqkv, qkv_bf, nullptr, M_TOT, 3 * D_MODEL, D_MODEL);

  rope_split<<<(BATCH * N_HEAD * SEQ * 64) / 256, 256, 0, stream>>>(
      qkv_bf, q_r, k_r, v_t);

  flash_attn<<<dim3(BATCH * N_HEAD, SEQ / 128), 256, 0, stream>>>(
      q_r, k_r, v_t, y_bf);

  // out = y @ Wproj + bproj  (4096 x 2048 x 2048), f32 output
  hgemm_bias<false><<<dim3(D_MODEL / 128, M_TOT / 128), 256, 0, stream>>>(
      y_bf, wproj_t, bproj, nullptr, out, M_TOT, D_MODEL, D_MODEL);
}
